// GTEProgramClassification_27986006900849
// MI455X (gfx1250) — matrix operand order repaired and checked by reading, not compiled, not run on hardware
//
#include <hip/hip_runtime.h>
#include <hip/hip_bf16.h>

typedef float  v8f   __attribute__((ext_vector_type(8)));
typedef __bf16 v16bf __attribute__((ext_vector_type(16)));
typedef __bf16 v8bf  __attribute__((ext_vector_type(8)));

#define D        256
#define LMAX     16
#define NCLS     104
#define NCLS_PAD 112
#define BSTR     264   // bf16 LDS row stride (elems); 528B = 16B-aligned, rotates banks

#define V8F_ZERO {0.f,0.f,0.f,0.f,0.f,0.f,0.f,0.f}

__device__ __forceinline__ v16bf make_v16(v8bf lo, v8bf hi) {
  v16bf r;
#pragma unroll
  for (int i = 0; i < 8; ++i) { r[i] = lo[i]; r[i + 8] = hi[i]; }
  return r;
}

// A fragment (16x32 bf16, row in LDS): lane holds row M=lane&15;
// lo chunk = K[k32 + (lane>=16?8:0) .. +7], hi chunk = +16 (per ISA A-layout table)
__device__ __forceinline__ v16bf load_a_lds(const __bf16* row, int k32, int lane) {
  const int k0 = k32 + ((lane >> 4) << 3);
  v8bf lo = *(const v8bf*)(row + k0);
  v8bf hi = *(const v8bf*)(row + k0 + 16);
  return make_v16(lo, hi);
}

// B fragment (32x16 bf16, W row in global): lane holds col N=lane&15;
// lo chunk = K[k32 + (lane>=16?8:0) .. +7], hi chunk = +16 (matches A layout)
__device__ __forceinline__ v16bf load_b_glb(const __bf16* row, int k32, int lane) {
  const int k0 = k32 + ((lane >> 4) << 3);
  v8bf lo = *(const v8bf*)(row + k0);
  v8bf hi = *(const v8bf*)(row + k0 + 16);
  return make_v16(lo, hi);
}

__global__ void convert_weights_kernel(const float* __restrict__ wih_f,
                                       const float* __restrict__ whh_f,
                                       const float* __restrict__ fcw_f,
                                       __bf16* __restrict__ wih,
                                       __bf16* __restrict__ whh,
                                       __bf16* __restrict__ fcw) {
  const int i = blockIdx.x * blockDim.x + threadIdx.x;
  if (i < 3 * D * D) { wih[i] = (__bf16)wih_f[i]; whh[i] = (__bf16)whh_f[i]; }
  if (i < NCLS_PAD * D) fcw[i] = (i < NCLS * D) ? (__bf16)fcw_f[i] : (__bf16)0.f;
}

__global__ __launch_bounds__(256)
void gte_fused_kernel(const int*   __restrict__ msg_tokens,
                      const int*   __restrict__ degrees,
                      const float* __restrict__ emb,
                      const __bf16* __restrict__ wih,
                      const __bf16* __restrict__ whh,
                      const __bf16* __restrict__ fcw,
                      const float* __restrict__ b_ih,
                      const float* __restrict__ b_hh,
                      const float* __restrict__ ln_g,
                      const float* __restrict__ ln_b,
                      const float* __restrict__ fc_b,
                      float* __restrict__ out) {
  __shared__ __align__(16) float  h01f[16 * D];    // h0, overwritten in place by h1
  __shared__ __align__(16) float  lastf[16 * D];
  __shared__ __align__(16) __bf16 h0b [16 * BSTR];
  __shared__ __align__(16) __bf16 lastb[16 * BSTR];
  __shared__ __align__(16) __bf16 ftb [16 * BSTR];
  __shared__ float stats[32];

  const int tid  = threadIdx.x;
  const int node = tid >> 4;        // 16 threads per node
  const int l16  = tid & 15;
  const long long gnode = (long long)blockIdx.x * 16 + node;

  // ---- Stage 1: mailbox gather; h0 = sum msgs[0..deg-2], last = msgs[deg-1]
  const int deg = degrees[gnode];
  float h0r[16], lar[16];
#pragma unroll
  for (int j = 0; j < 16; ++j) { h0r[j] = 0.f; lar[j] = 0.f; }
  const int* toks = msg_tokens + gnode * LMAX;
  for (int l = 0; l < deg; ++l) {
    const float* row = emb + (long long)toks[l] * D;
    const bool isLast = (l == deg - 1);
#pragma unroll
    for (int j = 0; j < 16; ++j) {
      const float v = row[l16 + j * 16];
      if (isLast) lar[j] = v; else h0r[j] += v;
    }
  }
#pragma unroll
  for (int j = 0; j < 16; ++j) {
    const int d = l16 + j * 16;
    h01f[node * D + d]     = h0r[j];
    lastf[node * D + d]    = lar[j];
    h0b [node * BSTR + d]  = (__bf16)h0r[j];
    lastb[node * BSTR + d] = (__bf16)lar[j];
  }
  __syncthreads();

  // ---- Stage 2: gi = last @ Wih^T, gh = h0 @ Whh^T via bf16 WMMA (f32 acc) + GRU gates
  const int wave = tid >> 5;
  const int lane = tid & 31;
  const int arow = lane & 15;
  const int mo   = (lane >> 4) << 3;   // accumulator row offset per half-wave

#pragma unroll 1
  for (int t = 0; t < 2; ++t) {
    const int dt = wave + t * 8;       // d-tile 0..15 (16 cols of D each)
    v8f air = V8F_ZERO, aiz = V8F_ZERO, ain = V8F_ZERO;
    v8f ahr = V8F_ZERO, ahz = V8F_ZERO, ahn = V8F_ZERO;
    const __bf16* wr_ir = wih + (long long)(dt * 16 + arow) * D;
    const __bf16* wr_iz = wr_ir + (long long)D * D;
    const __bf16* wr_in = wr_iz + (long long)D * D;
    const __bf16* wr_hr = whh + (long long)(dt * 16 + arow) * D;
    const __bf16* wr_hz = wr_hr + (long long)D * D;
    const __bf16* wr_hn = wr_hz + (long long)D * D;
    const __bf16* aL = lastb + arow * BSTR;
    const __bf16* aH = h0b  + arow * BSTR;
#pragma unroll
    for (int kb = 0; kb < 8; ++kb) {
      const int k32 = kb * 32;
      const v16bf fa_l = load_a_lds(aL, k32, lane);
      const v16bf fa_h = load_a_lds(aH, k32, lane);
      v16bf fb;
      fb  = load_b_glb(wr_ir, k32, lane);
      air = __builtin_amdgcn_wmma_f32_16x16x32_bf16(false, fa_l, false, fb, (short)0, air, false, false);
      fb  = load_b_glb(wr_iz, k32, lane);
      aiz = __builtin_amdgcn_wmma_f32_16x16x32_bf16(false, fa_l, false, fb, (short)0, aiz, false, false);
      fb  = load_b_glb(wr_in, k32, lane);
      ain = __builtin_amdgcn_wmma_f32_16x16x32_bf16(false, fa_l, false, fb, (short)0, ain, false, false);
      fb  = load_b_glb(wr_hr, k32, lane);
      ahr = __builtin_amdgcn_wmma_f32_16x16x32_bf16(false, fa_h, false, fb, (short)0, ahr, false, false);
      fb  = load_b_glb(wr_hz, k32, lane);
      ahz = __builtin_amdgcn_wmma_f32_16x16x32_bf16(false, fa_h, false, fb, (short)0, ahz, false, false);
      fb  = load_b_glb(wr_hn, k32, lane);
      ahn = __builtin_amdgcn_wmma_f32_16x16x32_bf16(false, fa_h, false, fb, (short)0, ahn, false, false);
    }
    // Gate math directly on accumulator layout: (M = v+mo, N = dt*16 + lane&15)
    const int d = dt * 16 + arow;
    const float bir = b_ih[d],         bhr = b_hh[d];
    const float biz = b_ih[D + d],     bhz = b_hh[D + d];
    const float bi_n = b_ih[2 * D + d], bh_n = b_hh[2 * D + d];
#pragma unroll
    for (int v = 0; v < 8; ++v) {
      const int m = v + mo;
      const float r  = 1.f / (1.f + __expf(-(air[v] + bir + ahr[v] + bhr)));
      const float z  = 1.f / (1.f + __expf(-(aiz[v] + biz + ahz[v] + bhz)));
      const float nn = tanhf(ain[v] + bi_n + r * (ahn[v] + bh_n));
      const float h0v = h01f[m * D + d];
      h01f[m * D + d] = (1.f - z) * nn + z * h0v;   // h1 written in place
    }
  }
  __syncthreads();

  // ---- Stage 3: LayerNorm (eps 1e-5) + degree==1 bypass, re-quantize ft -> bf16
  float s = 0.f, s2 = 0.f;
#pragma unroll
  for (int j = 0; j < 16; ++j) {
    const float v = h01f[node * D + l16 + j * 16];
    s += v; s2 += v * v;
  }
#pragma unroll
  for (int o = 8; o >= 1; o >>= 1) {
    s  += __shfl_xor(s,  o, 16);
    s2 += __shfl_xor(s2, o, 16);
  }
  if (l16 == 0) {
    const float mu  = s * (1.f / D);
    const float var = s2 * (1.f / D) - mu * mu;
    stats[node * 2]     = mu;
    stats[node * 2 + 1] = rsqrtf(var + 1e-5f);
  }
  __syncthreads();
  {
    const float mu = stats[node * 2], inv = stats[node * 2 + 1];
    const bool bypass = (deg == 1);
#pragma unroll
    for (int j = 0; j < 16; ++j) {
      const int d = l16 + j * 16;
      const float ftv = bypass ? lastf[node * D + d]
                               : (h01f[node * D + d] - mu) * inv * ln_g[d] + ln_b[d];
      ftb[node * BSTR + d] = (__bf16)ftv;
    }
  }
  __syncthreads();

  // ---- Stage 4: logits = ft @ fc_w^T + fc_b  (7 waves x one 16x16 tile; rows 104..111 zero-padded)
  if (wave < 7) {
    v8f acc = V8F_ZERO;
    const __bf16* fr = fcw + (long long)(wave * 16 + arow) * D;
    const __bf16* aF = ftb + arow * BSTR;
#pragma unroll
    for (int kb = 0; kb < 8; ++kb) {
      const int k32 = kb * 32;
      const v16bf fa = load_a_lds(aF, k32, lane);
      const v16bf fb = load_b_glb(fr, k32, lane);
      acc = __builtin_amdgcn_wmma_f32_16x16x32_bf16(false, fa, false, fb, (short)0, acc, false, false);
    }
    const int c = wave * 16 + arow;
    if (c < NCLS) {
      const float bc = fc_b[c];
      const long long obase = (long long)blockIdx.x * 16;
#pragma unroll
      for (int v = 0; v < 8; ++v) {
        out[(obase + v + mo) * NCLS + c] = acc[v] + bc;
      }
    }
  }
}

extern "C" void kernel_launch(void* const* d_in, const int* in_sizes, int n_in,
                              void* d_out, int out_size, void* d_ws, size_t ws_size,
                              hipStream_t stream) {
  const int*   msg_tokens = (const int*)d_in[0];
  const int*   degrees    = (const int*)d_in[1];
  const float* emb        = (const float*)d_in[2];
  const float* W_ih       = (const float*)d_in[3];
  const float* W_hh       = (const float*)d_in[4];
  const float* b_ih       = (const float*)d_in[5];
  const float* b_hh       = (const float*)d_in[6];
  const float* ln_g       = (const float*)d_in[7];
  const float* ln_b       = (const float*)d_in[8];
  const float* fc_w       = (const float*)d_in[9];
  const float* fc_b       = (const float*)d_in[10];
  float* out = (float*)d_out;

  // Workspace: bf16 weight copies (Wih 384KB, Whh 384KB, fc_w padded 56KB)
  __bf16* wih = (__bf16*)d_ws;
  __bf16* whh = wih + 3 * D * D;
  __bf16* fcw = whh + 3 * D * D;

  convert_weights_kernel<<<(3 * D * D + 255) / 256, 256, 0, stream>>>(
      W_ih, W_hh, fc_w, wih, whh, fcw);

  const int N = in_sizes[1];          // N_DST (50000, divisible by 16)
  const int nblocks = N / 16;
  gte_fused_kernel<<<nblocks, 256, 0, stream>>>(
      msg_tokens, degrees, emb, wih, whh, fcw,
      b_ih, b_hh, ln_g, ln_b, fc_b, out);
}
